// SwinTransformerBlock3D_17386027614783
// MI455X (gfx1250) — matrix operand order repaired and checked by reading, not compiled, not run on hardware
//
#include <hip/hip_runtime.h>
#include <hip/hip_bf16.h>

// ---------------------------------------------------------------------------
// Swin-3D block, MI455X (gfx1250, wave32). All GEMMs via v_wmma_f32_16x16x32_f16;
// attention K/V staged into LDS by the Tensor Data Mover (tensor_load_to_lds).
// ---------------------------------------------------------------------------

typedef __attribute__((ext_vector_type(16))) _Float16 v16h;
typedef __attribute__((ext_vector_type(8)))  _Float16 v8h;
typedef __attribute__((ext_vector_type(8)))  float    v8f;
typedef __attribute__((ext_vector_type(4)))  unsigned int u32x4;
typedef __attribute__((ext_vector_type(8)))  int      i32x8;
typedef __attribute__((ext_vector_type(4)))  int      i32x4;

static constexpr int kH = 56, kW = 56, kD = 28;
static constexpr int kC = 96, kNH = 3, kHD = 32, kFF = 384;
static constexpr int kN = 343, kNp = 352;                 // tokens/window, padded
static constexpr int kNW = 256;                           // 8*8*4 windows
static constexpr int kL  = kH * kW * kD;                  // 87808 = 16*5488
static constexpr int kLw = kNW * kNp;                     // 90112 = 16*5632
static constexpr float kScale = 0.17677669529663687f;     // 32^-0.5

// ----------------------------- WMMA helpers --------------------------------
// A fragment (16x32 f16, M x K): lane m=lane&15, half=lane>>4.
// elements j<8  -> K = half*8 + j ; j>=8 -> K = 16 + half*8 + (j-8).
__device__ __forceinline__ v16h load_a_frag(const _Float16* base, int ld, int k0) {
  const int lane = threadIdx.x & 31;
  const _Float16* p = base + (size_t)(lane & 15) * ld + k0 + (lane >> 4) * 8;
  v8h lo = *(const v8h*)(p);
  v8h hi = *(const v8h*)(p + 16);
  v16h r;
#pragma unroll
  for (int i = 0; i < 8; ++i) { r[i] = lo[i]; r[i + 8] = hi[i]; }
  return r;
}

// B fragment (32x16 f16, K x N) from K-contiguous transposed storage Bt[n][k]:
// lane n=lane&15, half=lane>>4, e<8 -> K = half*8 + e ; e>=8 -> K = 16 + half*8 + (e-8).
__device__ __forceinline__ v16h load_b_frag(const _Float16* base, int ld, int k0) {
  const int lane = threadIdx.x & 31;
  const _Float16* p = base + (size_t)(lane & 15) * ld + k0 + (lane >> 4) * 8;
  v8h lo = *(const v8h*)(p);
  v8h hi = *(const v8h*)(p + 16);
  v16h r;
#pragma unroll
  for (int i = 0; i < 8; ++i) { r[i] = lo[i]; r[i + 8] = hi[i]; }
  return r;
}

__device__ __forceinline__ v8f wmma16(v16h a, v16h b, v8f c) {
  return __builtin_amdgcn_wmma_f32_16x16x32_f16(false, a, false, b, (short)0, c,
                                                false, false);
}
__device__ __forceinline__ v8f zero8() {
  v8f c = {0.f, 0.f, 0.f, 0.f, 0.f, 0.f, 0.f, 0.f};
  return c;
}
// C/D frag: VGPR r, lanes0-15: (M=r, N=lane); lanes16-31: (M=r+8, N=lane-16).

// --------------------- TDM: build D# and issue DMA to LDS ------------------
// D# group0/group1 per CDNA5 ISA ch.8: count=1, type=2(image), data_size=8B.
// 2D tile: tile_d0 qwords/row, tile_d1 rows, row stride stride0 qwords.
// 1D copy: tile_d1 = 0 (unused).
__device__ __forceinline__ void tdm_load_to_lds(unsigned lds_off, const void* gptr,
                                                unsigned tile_d0, unsigned tile_d1,
                                                unsigned tensor_d0, unsigned tensor_d1,
                                                unsigned long long stride0) {
  unsigned long long ga = (unsigned long long)(uintptr_t)gptr;
  u32x4 g0;
  g0[0] = 1u;                                                 // count=1, user D#
  g0[1] = lds_off;                                            // lds_addr (bytes)
  g0[2] = (unsigned)ga;                                       // global_addr[31:0]
  g0[3] = (unsigned)((ga >> 32) & 0x01FFFFFFu) | (2u << 30);  // addr[56:32]|type=2
  i32x8 g1;
  g1[0] = (int)(3u << 16);                                    // data_size = 8B
  g1[1] = (int)((tensor_d0 & 0xFFFFu) << 16);                 // tensor_dim0 lo
  g1[2] = (int)((tensor_d0 >> 16) | ((tensor_d1 & 0xFFFFu) << 16));
  g1[3] = (int)((tensor_d1 >> 16) | (tile_d0 << 16));         // tile_dim0
  g1[4] = (int)tile_d1;                                       // tile_dim1 (tile_dim2=0)
  g1[5] = (int)(unsigned)stride0;                             // tensor_dim0_stride lo
  g1[6] = (int)(unsigned)(stride0 >> 32);                     // stride hi (stride1=0)
  g1[7] = 0;
  i32x4 gz = {0, 0, 0, 0};
#if __clang_major__ >= 23
  i32x8 gz8 = {0, 0, 0, 0, 0, 0, 0, 0};
  __builtin_amdgcn_tensor_load_to_lds(g0, g1, gz, gz, gz8, 0);
#else
  __builtin_amdgcn_tensor_load_to_lds(g0, g1, gz, gz, 0);
#endif
}

// ------------------------- 0: weight transpose -> f16 ----------------------
__global__ void k_prep(const float* qkv_w, const float* proj_w,
                       const float* fc1_w, const float* fc2_w,
                       _Float16* qkvT, _Float16* projT,
                       _Float16* fc1T, _Float16* fc2T) {
  int i = blockIdx.x * 256 + threadIdx.x;
  const int S0 = 288 * 96, S1 = 96 * 96, S2 = 384 * 96, S3 = 96 * 384;
  if (i < S0) { int n = i / 96, k = i % 96; qkvT[i] = (_Float16)qkv_w[k * 288 + n]; return; }
  i -= S0;
  if (i < S1) { int n = i / 96, k = i % 96; projT[i] = (_Float16)proj_w[k * 96 + n]; return; }
  i -= S1;
  if (i < S2) { int n = i / 96, k = i % 96; fc1T[i] = (_Float16)fc1_w[k * 384 + n]; return; }
  i -= S2;
  if (i < S3) { int n = i / 384, k = i % 384; fc2T[i] = (_Float16)fc2_w[k * 96 + n]; return; }
}

// -------------------- 1: relative-position bias table ----------------------
// biasT[h][i][j], i,j in [0,352); pad cols get -65504 (exp underflows to 0).
__global__ void k_bias(const float* rpb, _Float16* biasT) {
  int e = blockIdx.x * 256 + threadIdx.x;
  if (e >= kNH * kNp * kNp) return;
  int h = e / (kNp * kNp), rem = e % (kNp * kNp);
  int i = rem / kNp, j = rem % kNp;
  float v;
  if (i < kN && j < kN) {
    int i0 = i / 49, i1 = (i / 7) % 7, i2 = i % 7;
    int j0 = j / 49, j1 = (j / 7) % 7, j2 = j % 7;
    int idx = (i0 - j0 + 6) * 169 + (i1 - j1 + 6) * 13 + (i2 - j2 + 6);
    v = rpb[idx * kNH + h];
  } else {
    v = -65504.0f;
  }
  biasT[e] = (_Float16)v;
}

// ------------- 2: LN1 + cyclic shift + window partition -> f16 -------------
__global__ void k_ln1(const float* x, const float* g, const float* b,
                      _Float16* xw) {
  int row = blockIdx.x * 256 + threadIdx.x;
  if (row >= kLw) return;
  _Float16* out = xw + (size_t)row * kC;
  int w = row / kNp, t = row % kNp;
  if (t >= kN) {
#pragma unroll 4
    for (int c = 0; c < kC; ++c) out[c] = (_Float16)0.f;
    return;
  }
  int wh = w >> 5, ww = (w >> 2) & 7, wd = w & 3;
  int th = t / 49, tw = (t / 7) % 7, td = t % 7;
  int ph = (wh * 7 + th + 3) % kH;
  int pw = (ww * 7 + tw + 3) % kW;
  int pd = (wd * 7 + td + 3) % kD;
  int tok = (ph * kW + pw) * kD + pd;
  const float* xr = x + (size_t)tok * kC;
  float m = 0.f;
#pragma unroll 4
  for (int c = 0; c < kC; ++c) m += xr[c];
  m *= (1.0f / kC);
  float var = 0.f;
#pragma unroll 4
  for (int c = 0; c < kC; ++c) { float d = xr[c] - m; var += d * d; }
  var *= (1.0f / kC);
  float rs = rsqrtf(var + 1e-5f);
#pragma unroll 4
  for (int c = 0; c < kC; ++c)
    out[c] = (_Float16)((xr[c] - m) * rs * g[c] + b[c]);
}

// ----------------------- 3: fused QKV GEMM (WMMA) --------------------------
// (90112 x 96) @ (96 x 288) + bias -> q (scaled, f16), k (f16), v transposed.
// Tile indices forced scalar: q/k/v selection becomes s_cbranch, div -> SALU.
__global__ void __launch_bounds__(256) k_qkv(const _Float16* A, const _Float16* Bt,
                                             const float* bias, _Float16* q,
                                             _Float16* kk, _Float16* vt) {
  const int TN = 288 / 16;
  int gw = __builtin_amdgcn_readfirstlane(blockIdx.x * 8 + (threadIdx.x >> 5));
  int tm = gw / TN, tn = gw % TN;
  if (tm >= kLw / 16) return;
  const _Float16* Ab = A + (size_t)(tm * 16) * kC;
  const _Float16* Bb = Bt + (size_t)(tn * 16) * kC;
  v8f c = zero8();
#pragma unroll
  for (int k0 = 0; k0 < kC; k0 += 32)
    c = wmma16(load_a_frag(Ab, kC, k0), load_b_frag(Bb, kC, k0), c);
  int lane = threadIdx.x & 31, nl = lane & 15, half = lane >> 4;
  int n = tn * 16 + nl;
  float bv = bias[n];
  if (tn < 6) {                         // Q, scaled
#pragma unroll
    for (int r = 0; r < 8; ++r)
      q[(size_t)(tm * 16 + r + half * 8) * kC + n] = (_Float16)((c[r] + bv) * kScale);
  } else if (tn < 12) {                 // K
#pragma unroll
    for (int r = 0; r < 8; ++r)
      kk[(size_t)(tm * 16 + r + half * 8) * kC + (n - 96)] = (_Float16)(c[r] + bv);
  } else {                              // V, stored transposed per (window, head)
    int head = (tn - 12) >> 1;                    // uniform
    int c2 = ((tn - 12) & 1) * 16 + nl;           // channel within head
    int w = tm / 22, tb = (tm % 22) * 16;         // 352 = 22*16 tiles per window
    _Float16* vrow = vt + ((size_t)(w * kNH + head) * kHD + c2) * kNp + tb;
#pragma unroll
    for (int r = 0; r < 8; ++r) vrow[r + half * 8] = (_Float16)(c[r] + bv);
  }
}

// ------- 4: fused attention per (window, head): QK^T+bias+mask+softmax+PV ---
__global__ void __launch_bounds__(256) k_attn(const _Float16* q, const _Float16* kk,
                                              const _Float16* vt,
                                              const _Float16* biasT, _Float16* o) {
  extern __shared__ _Float16 smem[];
  _Float16* sK = smem;                   // [352][32]  (token-major = Bt for QK^T)
  _Float16* sV = sK + kNp * kHD;         // [32][352]  (chan-major  = Bt for PV)
  _Float16* sP = sV + kHD * kNp;         // 8 waves x [16][352] prob panels

  int w = blockIdx.x / kNH, head = blockIdx.x % kNH;
  // --- TDM staging: one DMA per operand, issued by wave 0 only -------------
  if ((threadIdx.x >> 5) == 0) {
    // K: 352 rows of 32 f16 (8 qwords), row stride 96 f16 (24 qwords)
    tdm_load_to_lds(/*lds*/ 0u,
                    kk + (size_t)(w * kNp) * kC + head * kHD,
                    /*tile_d0*/ 8u, /*tile_d1*/ (unsigned)kNp,
                    /*tensor_d0*/ 8u, /*tensor_d1*/ (unsigned)kNp,
                    /*stride0*/ 24ull);
    // V^T: contiguous 32*352 f16 = 2816 qwords, 1D
    tdm_load_to_lds(/*lds*/ (unsigned)(kNp * kHD * 2),
                    vt + (size_t)(w * kNH + head) * kHD * kNp,
                    /*tile_d0*/ 2816u, /*tile_d1*/ 0u,
                    /*tensor_d0*/ 2816u, /*tensor_d1*/ 1u,
                    /*stride0*/ 2816ull);
    __builtin_amdgcn_s_wait_tensorcnt(0);
  }
  __syncthreads();

  int wave = threadIdx.x >> 5, lane = threadIdx.x & 31;
  int nl = lane & 15, half = lane >> 4;
  int wh = w >> 5, ww = (w >> 2) & 7, wd = w & 3;
  bool lastH = (wh == 7), lastW = (ww == 7), lastD = (wd == 3);
  _Float16* myP = sP + (size_t)wave * 16 * kNp;
  const _Float16* bh = biasT + (size_t)head * kNp * kNp;

  for (int mt = wave; mt < kNp / 16; mt += 8) {
    const _Float16* Aq = q + (size_t)(w * kNp + mt * 16) * kC + head * kHD;
    v16h a = load_a_frag(Aq, kC, 0);  // full K=32 in one fragment
    int gi[8];
    float rm[8], rs[8];
#pragma unroll
    for (int r = 0; r < 8; ++r) {
      int i = mt * 16 + r + half * 8;
      int th = i / 49, tw = (i / 7) % 7, td = i % 7;
      gi[r] = (lastH && th >= 4 ? 1 : 0) | (lastW && tw >= 4 ? 2 : 0) |
              (lastD && td >= 4 ? 4 : 0);
      rm[r] = -3.0e38f;
      rs[r] = 0.f;
    }
    // pass 1: row max over all score tiles
    for (int jt = 0; jt < kNp / 16; ++jt) {
      v16h b = load_b_frag(sK + (size_t)(jt * 16) * kHD, kHD, 0);
      v8f c = wmma16(a, b, zero8());
      int j = jt * 16 + nl;
      int jh = j / 49, jw = (j / 7) % 7, jd = j % 7;
      int gj = (lastH && jh >= 4 ? 1 : 0) | (lastW && jw >= 4 ? 2 : 0) |
               (lastD && jd >= 4 ? 4 : 0);
#pragma unroll
      for (int r = 0; r < 8; ++r) {
        int i = mt * 16 + r + half * 8;
        float s = c[r] + (float)bh[(size_t)i * kNp + j] +
                  (gi[r] == gj ? 0.f : -100.f);
        rm[r] = fmaxf(rm[r], s);
      }
    }
#pragma unroll
    for (int r = 0; r < 8; ++r)
      for (int m = 1; m < 16; m <<= 1)
        rm[r] = fmaxf(rm[r], __shfl_xor(rm[r], m, 32));
    // pass 2: recompute scores, exponentiate, accumulate row sums, stash f16 P
    for (int jt = 0; jt < kNp / 16; ++jt) {
      v16h b = load_b_frag(sK + (size_t)(jt * 16) * kHD, kHD, 0);
      v8f c = wmma16(a, b, zero8());
      int j = jt * 16 + nl;
      int jh = j / 49, jw = (j / 7) % 7, jd = j % 7;
      int gj = (lastH && jh >= 4 ? 1 : 0) | (lastW && jw >= 4 ? 2 : 0) |
               (lastD && jd >= 4 ? 4 : 0);
#pragma unroll
      for (int r = 0; r < 8; ++r) {
        int i = mt * 16 + r + half * 8;
        float s = c[r] + (float)bh[(size_t)i * kNp + j] +
                  (gi[r] == gj ? 0.f : -100.f);
        float p = __expf(s - rm[r]);
        rs[r] += p;
        myP[(size_t)(r + half * 8) * kNp + j] = (_Float16)p;
      }
    }
#pragma unroll
    for (int r = 0; r < 8; ++r)
      for (int m = 1; m < 16; m <<= 1) rs[r] += __shfl_xor(rs[r], m, 32);
    // panel is private to this wave; drain DS stores before fragment reads
    asm volatile("s_wait_dscnt 0x0" ::: "memory");
    // PV: (16 x 352) @ (352 x 32) in 11 K-steps, two N-tiles of 16 channels
    v8f c0 = zero8(), c1 = zero8();
    for (int kt = 0; kt < kNp / 32; ++kt) {
      v16h ap = load_a_frag(myP, kNp, kt * 32);
      v16h b0 = load_b_frag(sV, kNp, kt * 32);
      v16h b1 = load_b_frag(sV + (size_t)16 * kNp, kNp, kt * 32);
      c0 = wmma16(ap, b0, c0);
      c1 = wmma16(ap, b1, c1);
    }
#pragma unroll
    for (int r = 0; r < 8; ++r) {
      float inv = 1.0f / rs[r];
      size_t row = (size_t)(w * kNp + mt * 16 + r + half * 8);
      o[row * kC + head * kHD + nl] = (_Float16)(c0[r] * inv);
      o[row * kC + head * kHD + 16 + nl] = (_Float16)(c1[r] * inv);
    }
  }
}

// ---- 5: proj GEMM + window-reverse scatter + residual -> x1 (f32) ----------
__global__ void __launch_bounds__(256) k_proj(const _Float16* A, const _Float16* Bt,
                                              const float* bias, const float* x,
                                              float* x1) {
  const int TN = 96 / 16;
  int gw = __builtin_amdgcn_readfirstlane(blockIdx.x * 8 + (threadIdx.x >> 5));
  int tm = gw / TN, tn = gw % TN;
  if (tm >= kLw / 16) return;
  const _Float16* Ab = A + (size_t)(tm * 16) * kC;
  const _Float16* Bb = Bt + (size_t)(tn * 16) * kC;
  v8f c = zero8();
#pragma unroll
  for (int k0 = 0; k0 < kC; k0 += 32)
    c = wmma16(load_a_frag(Ab, kC, k0), load_b_frag(Bb, kC, k0), c);
  int lane = threadIdx.x & 31, nl = lane & 15, half = lane >> 4;
  int n = tn * 16 + nl;
  float bv = bias[n];
  int w = tm / 22, tb = (tm % 22) * 16;           // scalar window / token base
  int wh = w >> 5, ww = (w >> 2) & 7, wd = w & 3; // uniform window coords
#pragma unroll
  for (int r = 0; r < 8; ++r) {
    int t = tb + r + half * 8;
    if (t >= kN) continue;  // padded window rows
    int th = t / 49, tw = (t / 7) % 7, td = t % 7;
    int ph = (wh * 7 + th + 3) % kH;
    int pw = (ww * 7 + tw + 3) % kW;
    int pd = (wd * 7 + td + 3) % kD;
    size_t tok = (size_t)(ph * kW + pw) * kD + pd;
    x1[tok * kC + n] = x[tok * kC + n] + c[r] + bv;
  }
}

// ----------------------------- 6: LN2 -> f16 -------------------------------
__global__ void k_ln2(const float* x1, const float* g, const float* b,
                      _Float16* h2) {
  int row = blockIdx.x * 256 + threadIdx.x;
  if (row >= kL) return;
  const float* xr = x1 + (size_t)row * kC;
  float m = 0.f;
#pragma unroll 4
  for (int c = 0; c < kC; ++c) m += xr[c];
  m *= (1.0f / kC);
  float var = 0.f;
#pragma unroll 4
  for (int c = 0; c < kC; ++c) { float d = xr[c] - m; var += d * d; }
  var *= (1.0f / kC);
  float rs = rsqrtf(var + 1e-5f);
  _Float16* out = h2 + (size_t)row * kC;
#pragma unroll 4
  for (int c = 0; c < kC; ++c)
    out[c] = (_Float16)((xr[c] - m) * rs * g[c] + b[c]);
}

// ------------------------- 7: fc1 GEMM + exact GELU ------------------------
__global__ void __launch_bounds__(256) k_fc1(const _Float16* A, const _Float16* Bt,
                                             const float* bias, _Float16* gbuf) {
  const int TN = kFF / 16;
  int gw = __builtin_amdgcn_readfirstlane(blockIdx.x * 8 + (threadIdx.x >> 5));
  int tm = gw / TN, tn = gw % TN;
  if (tm >= kL / 16) return;
  const _Float16* Ab = A + (size_t)(tm * 16) * kC;
  const _Float16* Bb = Bt + (size_t)(tn * 16) * kC;
  v8f c = zero8();
#pragma unroll
  for (int k0 = 0; k0 < kC; k0 += 32)
    c = wmma16(load_a_frag(Ab, kC, k0), load_b_frag(Bb, kC, k0), c);
  int lane = threadIdx.x & 31, nl = lane & 15, half = lane >> 4;
  int n = tn * 16 + nl;
  float bv = bias[n];
#pragma unroll
  for (int r = 0; r < 8; ++r) {
    size_t row = (size_t)(tm * 16 + r + half * 8);
    float v = c[r] + bv;
    float gl = 0.5f * v * (1.0f + erff(v * 0.70710678118654752f));
    gbuf[row * kFF + n] = (_Float16)gl;
  }
}

// ----------------------- 8: fc2 GEMM + residual -> out ---------------------
__global__ void __launch_bounds__(256) k_fc2(const _Float16* A, const _Float16* Bt,
                                             const float* bias, const float* x1,
                                             float* out) {
  const int TN = 96 / 16;
  int gw = __builtin_amdgcn_readfirstlane(blockIdx.x * 8 + (threadIdx.x >> 5));
  int tm = gw / TN, tn = gw % TN;
  if (tm >= kL / 16) return;
  const _Float16* Ab = A + (size_t)(tm * 16) * kFF;
  const _Float16* Bb = Bt + (size_t)(tn * 16) * kFF;
  v8f c = zero8();
#pragma unroll
  for (int k0 = 0; k0 < kFF; k0 += 32)
    c = wmma16(load_a_frag(Ab, kFF, k0), load_b_frag(Bb, kFF, k0), c);
  int lane = threadIdx.x & 31, nl = lane & 15, half = lane >> 4;
  int n = tn * 16 + nl;
  float bv = bias[n];
#pragma unroll
  for (int r = 0; r < 8; ++r) {
    size_t row = (size_t)(tm * 16 + r + half * 8);
    out[row * kC + n] = x1[row * kC + n] + c[r] + bv;
  }
}

// ---------------------------------------------------------------------------
extern "C" void kernel_launch(void* const* d_in, const int* in_sizes, int n_in,
                              void* d_out, int out_size, void* d_ws,
                              size_t ws_size, hipStream_t stream) {
  const float* x      = (const float*)d_in[0];
  const float* n1g    = (const float*)d_in[1];
  const float* n1b    = (const float*)d_in[2];
  const float* qkv_w  = (const float*)d_in[3];
  const float* qkv_b  = (const float*)d_in[4];
  const float* rpb    = (const float*)d_in[5];
  const float* proj_w = (const float*)d_in[6];
  const float* proj_b = (const float*)d_in[7];
  const float* n2g    = (const float*)d_in[8];
  const float* n2b    = (const float*)d_in[9];
  const float* fc1_w  = (const float*)d_in[10];
  const float* fc1_b  = (const float*)d_in[11];
  const float* fc2_w  = (const float*)d_in[12];
  const float* fc2_b  = (const float*)d_in[13];
  float* out = (float*)d_out;

  char* ws = (char*)d_ws;
  size_t off = 0;
  auto alloc = [&](size_t bytes) {
    void* p = ws + off;
    off = (off + bytes + 255) & ~(size_t)255;
    return p;
  };
  _Float16* xw    = (_Float16*)alloc((size_t)kLw * kC * 2);
  _Float16* qb    = (_Float16*)alloc((size_t)kLw * kC * 2);
  _Float16* kb    = (_Float16*)alloc((size_t)kLw * kC * 2);
  _Float16* vt    = (_Float16*)alloc((size_t)kNW * kNH * kHD * kNp * 2);
  _Float16* ob    = (_Float16*)alloc((size_t)kLw * kC * 2);
  float*    x1    = (float*)alloc((size_t)kL * kC * 4);
  _Float16* h2    = (_Float16*)alloc((size_t)kL * kC * 2);
  _Float16* gbuf  = (_Float16*)alloc((size_t)kL * kFF * 2);
  _Float16* biasT = (_Float16*)alloc((size_t)kNH * kNp * kNp * 2);
  _Float16* qkvT  = (_Float16*)alloc((size_t)288 * 96 * 2);
  _Float16* projT = (_Float16*)alloc((size_t)96 * 96 * 2);
  _Float16* fc1T  = (_Float16*)alloc((size_t)384 * 96 * 2);
  _Float16* fc2T  = (_Float16*)alloc((size_t)96 * 384 * 2);

  const int attn_lds = (kNp * kHD + kHD * kNp + 8 * 16 * kNp) * 2;  // 135168 B
  (void)hipFuncSetAttribute((const void*)k_attn,
                            hipFuncAttributeMaxDynamicSharedMemorySize,
                            attn_lds);

  k_prep<<<dim3(432), dim3(256), 0, stream>>>(qkv_w, proj_w, fc1_w, fc2_w,
                                              qkvT, projT, fc1T, fc2T);
  k_bias<<<dim3((kNH * kNp * kNp + 255) / 256), dim3(256), 0, stream>>>(rpb, biasT);
  k_ln1<<<dim3(kLw / 256), dim3(256), 0, stream>>>(x, n1g, n1b, xw);
  k_qkv<<<dim3((kLw / 16) * (288 / 16) / 8), dim3(256), 0, stream>>>(
      xw, qkvT, qkv_b, qb, kb, vt);
  k_attn<<<dim3(kNW * kNH), dim3(256), attn_lds, stream>>>(qb, kb, vt, biasT, ob);
  k_proj<<<dim3((kLw / 16) * (96 / 16) / 8), dim3(256), 0, stream>>>(
      ob, projT, proj_b, x, x1);
  k_ln2<<<dim3((kL + 255) / 256), dim3(256), 0, stream>>>(x1, n2g, n2b, h2);
  k_fc1<<<dim3((kL / 16) * (kFF / 16) / 8), dim3(256), 0, stream>>>(
      h2, fc1T, fc1_b, gbuf);
  k_fc2<<<dim3((kL / 16) * (96 / 16) / 8), dim3(256), 0, stream>>>(
      gbuf, fc2T, fc2_b, x1, out);
}
